// CreateNetFusion_16303695856229
// MI455X (gfx1250) — compile-verified, matrix operand order checked by reading, not run on hardware
//
#include <hip/hip_runtime.h>

typedef _Float16 h16;
typedef __attribute__((ext_vector_type(16))) _Float16 v16h;
typedef __attribute__((ext_vector_type(8)))  float    v8f;
typedef __attribute__((ext_vector_type(4)))  float    vf4;

#define WAVES_PER_BLOCK 8
#define GM 4   // M subtiles per wave in 1x1 GEMM
#define GN 2   // N subtiles per wave in 1x1 GEMM
#define CM 2   // M subtiles per wave in 3x3 conv
#define CN 2   // N subtiles per wave in 3x3 conv

// ---------------------------------------------------------------------------
// WMMA fragment loader: 16-bit 16x32 A/B fragment, one K-slice of 32.
// Lanes 0-15 hold K={kb..kb+7, kb+16..kb+23}; lanes 16-31 K={kb+8..kb+15, kb+24..kb+31}.
// kb is a compile-time constant in the templated kernels -> both b128 loads
// become immediate-offset loads off one per-row base address.
// ---------------------------------------------------------------------------
__device__ __forceinline__ v16h load_frag_k32(const h16* __restrict__ row, int kb, int lane) {
  const int lo = (lane & 16) ? 8 : 0;
  union { v16h v; vf4 f[2]; } u;
  u.f[0] = *reinterpret_cast<const vf4*>(row + kb + lo);
  u.f[1] = *reinterpret_cast<const vf4*>(row + kb + lo + 16);
  return u.v;
}

// ---------------------------------------------------------------------------
// 1x1 conv as GEMM, register-tiled 4Mx2N per wave, templated on CinPad and
// output channel stride so the K-loop fully unrolls and all loads/stores use
// immediate offsets. 12 b128 loads feed 8 WMMAs per K-step (~21 FLOP/B).
// ---------------------------------------------------------------------------
template<int CINPAD, int OUTSTRIDE>
__global__ __launch_bounds__(256) void gemm1x1_t(
    const h16* __restrict__ in, const h16* __restrict__ w, const float* __restrict__ bias,
    h16* __restrict__ out, int P, int Cout, int CoutPad, int Coff)
{
  const int lane = threadIdx.x & 31;
  const int wave = threadIdx.x >> 5;
  const int tilesN = CoutPad >> 5;                 // 32-wide N groups
  const long tilesM = (long)((P + 63) >> 6);       // 64-wide M groups
  const long tile = (long)blockIdx.x * WAVES_PER_BLOCK + wave;
  if (tile >= tilesM * tilesN) return;
  const int tM = (int)(tile / tilesN);
  const int tN = (int)(tile % tilesN);

  const h16* arow[GM];
#pragma unroll
  for (int i = 0; i < GM; ++i) {
    int m = tM * 64 + i * 16 + (lane & 15);
    if (m >= P) m = P - 1;                         // clamp; stores masked later
    arow[i] = in + (size_t)m * CINPAD;
  }
  int cobase[GN];
  const h16* brow[GN];
#pragma unroll
  for (int j = 0; j < GN; ++j) {
    cobase[j] = tN * 32 + j * 16 + (lane & 15);
    brow[j] = w + (size_t)cobase[j] * CINPAD;
  }

  const v8f vzero = {};
  v8f acc[GM][GN];
#pragma unroll
  for (int i = 0; i < GM; ++i)
#pragma unroll
    for (int j = 0; j < GN; ++j) acc[i][j] = vzero;

#pragma unroll
  for (int kb = 0; kb < CINPAD; kb += 32) {
    v16h a[GM], b[GN];
#pragma unroll
    for (int i = 0; i < GM; ++i) a[i] = load_frag_k32(arow[i], kb, lane);
#pragma unroll
    for (int j = 0; j < GN; ++j) b[j] = load_frag_k32(brow[j], kb, lane);
#pragma unroll
    for (int i = 0; i < GM; ++i)
#pragma unroll
      for (int j = 0; j < GN; ++j)
        acc[i][j] = __builtin_amdgcn_wmma_f32_16x16x32_f16(
            false, a[i], false, b[j], (short)0, acc[i][j], false, false);
  }

  const int moff = (lane & 16) ? 8 : 0;
  const bool full = (tM * 64 + 64 <= P);           // wave-uniform fast path
#pragma unroll
  for (int i = 0; i < GM; ++i) {
    const int m0 = tM * 64 + i * 16 + moff;
#pragma unroll
    for (int j = 0; j < GN; ++j) {
      const int co = cobase[j];
      const float bv = (co < Cout) ? bias[co] : 0.0f;
      union { v8f v; float f[8]; } r; r.v = acc[i][j];
      h16* po = out + (size_t)m0 * OUTSTRIDE + Coff + co;
      if (full) {
#pragma unroll
        for (int k = 0; k < 8; ++k)
          po[(size_t)k * OUTSTRIDE] = (h16)fmaxf(r.f[k] + bv, 0.0f);  // imm offsets
      } else {
#pragma unroll
        for (int k = 0; k < 8; ++k)
          if (m0 + k < P) po[(size_t)k * OUTSTRIDE] = (h16)fmaxf(r.f[k] + bv, 0.0f);
      }
    }
  }
}

// ---------------------------------------------------------------------------
// 3x3 same-pad conv as implicit GEMM, register-tiled 2Mx2N per wave.
// Tap loop and K-loop fully unroll (CINPAD is 32 or 64); weight fragment
// addresses fold into immediate offsets. w: [CoutPad][9][CINPAD] f16.
// ---------------------------------------------------------------------------
template<int CINPAD, int OUTSTRIDE>
__global__ __launch_bounds__(256) void conv3x3_t(
    const h16* __restrict__ in, const h16* __restrict__ w, const float* __restrict__ bias,
    h16* __restrict__ out, int N, int H, int W, int Cout, int CoutPad, int Coff)
{
  const int lane = threadIdx.x & 31;
  const int wave = threadIdx.x >> 5;
  const int P = N * H * W;
  const int tilesN = CoutPad >> 5;
  const long tilesM = (long)((P + 31) >> 5);
  const long tile = (long)blockIdx.x * WAVES_PER_BLOCK + wave;
  if (tile >= tilesM * tilesN) return;
  const int tM = (int)(tile / tilesN);
  const int tN = (int)(tile % tilesN);

  int pn[CM], py[CM], px[CM];
#pragma unroll
  for (int i = 0; i < CM; ++i) {
    int m = tM * 32 + i * 16 + (lane & 15);
    if (m >= P) m = P - 1;
    pn[i] = m / (H * W);
    const int rem = m % (H * W);
    py[i] = rem / W;
    px[i] = rem % W;
  }
  int cobase[CN];
  const h16* brow[CN];
#pragma unroll
  for (int j = 0; j < CN; ++j) {
    cobase[j] = tN * 32 + j * 16 + (lane & 15);
    brow[j] = w + (size_t)cobase[j] * 9 * CINPAD;
  }

  const v8f vzero = {};
  v8f acc[CM][CN];
#pragma unroll
  for (int i = 0; i < CM; ++i)
#pragma unroll
    for (int j = 0; j < CN; ++j) acc[i][j] = vzero;

  const v16h fzero = {};
#pragma unroll
  for (int t = 0; t < 9; ++t) {
    const int dy = t / 3 - 1, dx = t % 3 - 1;
    const h16* arow[CM];
    bool inb[CM];
#pragma unroll
    for (int i = 0; i < CM; ++i) {
      const int yy = py[i] + dy, xx = px[i] + dx;
      inb[i] = (yy >= 0) && (yy < H) && (xx >= 0) && (xx < W);
      arow[i] = in + (((size_t)pn[i] * H + (size_t)(inb[i] ? yy : py[i])) * W +
                      (inb[i] ? xx : px[i])) * CINPAD;
    }
#pragma unroll
    for (int kb = 0; kb < CINPAD; kb += 32) {
      v16h a[CM], b[CN];
#pragma unroll
      for (int i = 0; i < CM; ++i) {
        a[i] = load_frag_k32(arow[i], kb, lane);
        if (!inb[i]) a[i] = fzero;
      }
#pragma unroll
      for (int j = 0; j < CN; ++j)
        b[j] = load_frag_k32(brow[j] + t * CINPAD, kb, lane);
#pragma unroll
      for (int i = 0; i < CM; ++i)
#pragma unroll
        for (int j = 0; j < CN; ++j)
          acc[i][j] = __builtin_amdgcn_wmma_f32_16x16x32_f16(
              false, a[i], false, b[j], (short)0, acc[i][j], false, false);
    }
  }

  const int moff = (lane & 16) ? 8 : 0;
  const bool full = (tM * 32 + 32 <= P);
#pragma unroll
  for (int i = 0; i < CM; ++i) {
    const int m0 = tM * 32 + i * 16 + moff;
#pragma unroll
    for (int j = 0; j < CN; ++j) {
      const int co = cobase[j];
      const float bv = (co < Cout) ? bias[co] : 0.0f;
      union { v8f v; float f[8]; } r; r.v = acc[i][j];
      h16* po = out + (size_t)m0 * OUTSTRIDE + Coff + co;
      if (full) {
#pragma unroll
        for (int k = 0; k < 8; ++k)
          po[(size_t)k * OUTSTRIDE] = (h16)fmaxf(r.f[k] + bv, 0.0f);
      } else {
#pragma unroll
        for (int k = 0; k < 8; ++k)
          if (m0 + k < P) po[(size_t)k * OUTSTRIDE] = (h16)fmaxf(r.f[k] + bv, 0.0f);
      }
    }
  }
}

// ---------------------------------------------------------------------------
// Stem: fused mask split + conv 3x3 s2 (Cin=3) + relu -> NHWC f16 [8][255][255][64]
// ---------------------------------------------------------------------------
__global__ __launch_bounds__(256) void conv0_stem(
    const float* __restrict__ x, const float* __restrict__ w, const float* __restrict__ bias,
    h16* __restrict__ out)
{
  const int HO = 255, WO = 255, HI = 512, WI = 512;
  const long total = 8L * HO * WO;
  const long idx = (long)blockIdx.x * blockDim.x + threadIdx.x;
  if (idx >= total) return;
  const int nb = (int)(idx / (HO * WO));
  const int rem = (int)(idx % (HO * WO));
  const int oy = rem / WO, ox = rem % WO;
  const int b = nb & 3;
  const bool fg = nb < 4;
  const float* xb = x + (size_t)b * 4 * HI * WI;
  float fin[27];
#pragma unroll
  for (int c = 0; c < 3; ++c)
#pragma unroll
    for (int ky = 0; ky < 3; ++ky)
#pragma unroll
      for (int kx = 0; kx < 3; ++kx) {
        const int iy = oy * 2 + ky, ix = ox * 2 + kx;
        const float img = xb[(size_t)c * HI * WI + (size_t)iy * WI + ix];
        const float mk  = xb[(size_t)3 * HI * WI + (size_t)iy * WI + ix];
        fin[(c * 3 + ky) * 3 + kx] = img * (fg ? mk : (1.0f - mk));
      }
  h16* orow = out + (size_t)idx * 64;
  for (int co = 0; co < 64; ++co) {
    const float* wc = w + co * 27;
    float s = bias[co];
#pragma unroll
    for (int t = 0; t < 27; ++t) s = fmaf(fin[t], wc[t], s);
    orow[co] = (h16)fmaxf(s, 0.0f);
  }
}

// ---------------------------------------------------------------------------
// MaxPool 3x3 stride 2 VALID, NHWC f16.
// ---------------------------------------------------------------------------
__global__ __launch_bounds__(256) void maxpool3x3s2(
    const h16* __restrict__ in, h16* __restrict__ out,
    int N, int H, int W, int C, int Ho, int Wo)
{
  const long total = (long)N * Ho * Wo * C;
  const long idx = (long)blockIdx.x * blockDim.x + threadIdx.x;
  if (idx >= total) return;
  const int c = (int)(idx % C);
  long r = idx / C;
  const int ox = (int)(r % Wo); r /= Wo;
  const int oy = (int)(r % Ho);
  const int n = (int)(r / Ho);
  float mx = -3.0e38f;
#pragma unroll
  for (int dy = 0; dy < 3; ++dy)
#pragma unroll
    for (int dx = 0; dx < 3; ++dx) {
      const int iy = oy * 2 + dy, ix = ox * 2 + dx;
      const float v = (float)in[(((size_t)n * H + iy) * W + ix) * C + c];
      mx = fmaxf(mx, v);
    }
  out[idx] = (h16)mx;
}

// ---------------------------------------------------------------------------
// Global average pool over HW: in [8][HW][C] f16 -> pooled [8][C] f32.
// ---------------------------------------------------------------------------
__global__ void avgpool_global(const h16* __restrict__ in, float* __restrict__ pooled,
                               int HW, int C)
{
  const int n = blockIdx.x;
  const int c = threadIdx.x;
  const h16* base = in + (size_t)n * HW * C + c;
  float s = 0.0f;
  for (int p = 0; p < HW; ++p) s += (float)base[(size_t)p * C];
  pooled[n * C + c] = s / (float)HW;
}

// ---------------------------------------------------------------------------
// param[b][o] = bias_f[o] + bias_b[o] + pooled[b]·Wf[o] + pooled[b+4]·Wb[o]
// ---------------------------------------------------------------------------
__global__ __launch_bounds__(256) void fc_pair(
    const float* __restrict__ pooled,
    const float* __restrict__ wf, const float* __restrict__ bf_,
    const float* __restrict__ wb, const float* __restrict__ bb_,
    float* __restrict__ param)
{
  const int idx = blockIdx.x * blockDim.x + threadIdx.x;
  if (idx >= 4 * 384) return;
  const int b = idx / 384, o = idx % 384;
  const float* pf = pooled + b * 512;
  const float* pb = pooled + (b + 4) * 512;
  float s = bf_[o] + bb_[o];
  for (int k = 0; k < 512; ++k)
    s = fmaf(pf[k], wf[(size_t)o * 512 + k], fmaf(pb[k], wb[(size_t)o * 512 + k], s));
  param[idx] = s;
}

// ---------------------------------------------------------------------------
// Curve filter: scan-of-clips == prefix[K0]/128 + (v - K0/128)*p[K0].
// Params staged into LDS with GLOBAL_LOAD_ASYNC_TO_LDS_B32 (GV mode,
// ASYNCcnt path), then prefix-summed. out = w1[c]*sc1*tot1 + w2[c]*sc2*tot2.
// ---------------------------------------------------------------------------
__global__ __launch_bounds__(256) void curve_apply(
    const float* __restrict__ x, const float* __restrict__ p1, const float* __restrict__ p2,
    const float* __restrict__ w1, const float* __restrict__ w2, float* __restrict__ out)
{
  const int c = blockIdx.y;
  const int b = blockIdx.z;
  __shared__ float q1[128], q2[128], s1[129], s2[129];
  const float* pp1 = p1 + ((size_t)b * 3 + c) * 128;
  const float* pp2 = p2 + ((size_t)b * 3 + c) * 128;

  // Async global->LDS staging of the 2x128 curve params (one b32 per lane).
  {
    float* dst;
    const float* src;
    if (threadIdx.x < 128) { dst = &q1[threadIdx.x];        src = pp1 + threadIdx.x; }
    else                   { dst = &q2[threadIdx.x - 128];  src = pp2 + (threadIdx.x - 128); }
    const unsigned lds_off = (unsigned)(size_t)dst;        // aperture: addr[31:0] = LDS offset
    const unsigned long long ga = (unsigned long long)(size_t)src;
    asm volatile("global_load_async_to_lds_b32 %0, %1, off"
                 :: "v"(lds_off), "v"(ga) : "memory");
    asm volatile("s_wait_asynccnt 0x0" ::: "memory");
  }
  __syncthreads();

  if (threadIdx.x == 0) { float a = 0.f; for (int k = 0; k < 128; ++k) { s1[k] = a; a += q1[k]; } s1[128] = a; }
  if (threadIdx.x == 1) { float a = 0.f; for (int k = 0; k < 128; ++k) { s2[k] = a; a += q2[k]; } s2[128] = a; }
  __syncthreads();

  const float inv = 1.0f / 128.0f;
  const float sc1 = 128.0f / (s1[128] + 1e-30f);
  const float sc2 = 128.0f / (s2[128] + 1e-30f);
  const float ww1 = w1[c], ww2 = w2[c];
  const float* img = x + ((size_t)b * 4 + c) * 512 * 512;
  float* op = out + ((size_t)b * 3 + c) * 512 * 512;
  const long npix = 512L * 512L;
  for (long pix = (long)blockIdx.x * blockDim.x + threadIdx.x; pix < npix;
       pix += (long)gridDim.x * blockDim.x) {
    const float v = img[pix];
    const int K0 = (int)floorf(v * 128.0f);
    float t1, t2;
    if (K0 < 0)        { t1 = 0.0f;            t2 = 0.0f; }
    else if (K0 > 127) { t1 = s1[128] * inv;   t2 = s2[128] * inv; }
    else {
      const float rfrac = v - (float)K0 * inv;
      t1 = s1[K0] * inv + rfrac * q1[K0];
      t2 = s2[K0] * inv + rfrac * q2[K0];
    }
    op[pix] = ww1 * (t1 * sc1) + ww2 * (t2 * sc2);
  }
}

// ---------------------------------------------------------------------------
// Weight repack: fp32 OIHW -> f16 padded, zero-filled beyond Cout/Cin.
// ---------------------------------------------------------------------------
__global__ __launch_bounds__(256) void repack_w1x1(
    const float* __restrict__ src, h16* __restrict__ dst,
    int Cout, int Cin, int CoPad, int CiPad)
{
  const long total = (long)CoPad * CiPad;
  const long idx = (long)blockIdx.x * blockDim.x + threadIdx.x;
  if (idx >= total) return;
  const int ci = (int)(idx % CiPad);
  const int co = (int)(idx / CiPad);
  const float v = (co < Cout && ci < Cin) ? src[(size_t)co * Cin + ci] : 0.0f;
  dst[idx] = (h16)v;
}

__global__ __launch_bounds__(256) void repack_w3x3(
    const float* __restrict__ src, h16* __restrict__ dst,
    int Cout, int Cin, int CoPad, int CiPad)
{
  const long total = (long)CoPad * 9 * CiPad;
  const long idx = (long)blockIdx.x * blockDim.x + threadIdx.x;
  if (idx >= total) return;
  const int ci = (int)(idx % CiPad);
  long r = idx / CiPad;
  const int t = (int)(r % 9);
  const int co = (int)(r / 9);
  const float v = (co < Cout && ci < Cin) ? src[((size_t)co * Cin + ci) * 9 + t] : 0.0f;
  dst[idx] = (h16)v;
}

// ---------------------------------------------------------------------------
// Template dispatchers
// ---------------------------------------------------------------------------
struct GemmArgs {
  const h16* in; const h16* w; const float* bias; h16* out;
  int P; int Cout; int CoutPad; int Coff;
};

template<int CI, int OS>
static void launch_gemm_t(const GemmArgs& g, hipStream_t s) {
  const long tiles = (long)((g.P + 63) / 64) * (g.CoutPad / 32);
  const int blocks = (int)((tiles + WAVES_PER_BLOCK - 1) / WAVES_PER_BLOCK);
  gemm1x1_t<CI, OS><<<blocks, 256, 0, s>>>(g.in, g.w, g.bias, g.out, g.P, g.Cout, g.CoutPad, g.Coff);
}

static void launch_gemm(int CinPad, int OutStride, const GemmArgs& g, hipStream_t s) {
  if      (CinPad ==  64 && OutStride ==  32) launch_gemm_t< 64,  32>(g, s);
  else if (CinPad == 128 && OutStride ==  32) launch_gemm_t<128,  32>(g, s);
  else if (CinPad == 256 && OutStride ==  32) launch_gemm_t<256,  32>(g, s);
  else if (CinPad == 256 && OutStride ==  64) launch_gemm_t<256,  64>(g, s);
  else if (CinPad == 384 && OutStride ==  64) launch_gemm_t<384,  64>(g, s);
  else if (CinPad ==  32 && OutStride == 128) launch_gemm_t< 32, 128>(g, s);
  else if (CinPad ==  32 && OutStride == 256) launch_gemm_t< 32, 256>(g, s);
  else if (CinPad ==  64 && OutStride == 384) launch_gemm_t< 64, 384>(g, s);
  else if (CinPad ==  64 && OutStride == 512) launch_gemm_t< 64, 512>(g, s);
}

struct ConvArgs {
  const h16* in; const h16* w; const float* bias; h16* out;
  int H; int W; int Cout; int CoutPad; int Coff;
};

template<int CI, int OS>
static void launch_conv_t(const ConvArgs& c, hipStream_t s) {
  const int P = 8 * c.H * c.W;
  const long tiles = (long)((P + 31) / 32) * (c.CoutPad / 32);
  const int blocks = (int)((tiles + WAVES_PER_BLOCK - 1) / WAVES_PER_BLOCK);
  conv3x3_t<CI, OS><<<blocks, 256, 0, s>>>(c.in, c.w, c.bias, c.out, 8, c.H, c.W, c.Cout, c.CoutPad, c.Coff);
}

static void launch_conv(int CinPad, int OutStride, const ConvArgs& c, hipStream_t s) {
  if      (CinPad == 32 && OutStride == 128) launch_conv_t<32, 128>(c, s);
  else if (CinPad == 32 && OutStride == 256) launch_conv_t<32, 256>(c, s);
  else if (CinPad == 64 && OutStride == 384) launch_conv_t<64, 384>(c, s);
  else if (CinPad == 64 && OutStride == 512) launch_conv_t<64, 512>(c, s);
}

// ---------------------------------------------------------------------------
// Host orchestration
// ---------------------------------------------------------------------------
extern "C" void kernel_launch(void* const* d_in, const int* in_sizes, int n_in,
                              void* d_out, int out_size, void* d_ws, size_t ws_size,
                              hipStream_t stream)
{
  (void)in_sizes; (void)n_in; (void)out_size;
  const float* x   = (const float*)d_in[0];
  const float* c0w = (const float*)d_in[1];
  const float* c0b = (const float*)d_in[2];

  struct FireCfg { int cin, sq, sqp, ex; };
  const FireCfg FC[7] = {
    { 64, 16, 32,  64}, {128, 16, 32,  64},
    {128, 32, 32, 128}, {256, 32, 32, 128},
    {256, 48, 64, 192}, {384, 48, 64, 192},
    {384, 64, 64, 256}
  };

  char* ws = (char*)d_ws;
  size_t off = 0;
  auto alloc = [&](size_t bytes) -> void* {
    void* p = ws + off;
    off += (bytes + 255) & ~(size_t)255;
    return p;
  };

  h16* bufA = (h16*)alloc(68000000);   // max: conv0 out 8*255*255*64*2 = 66.6 MB
  h16* bufB = (h16*)alloc(34000000);   // max: f4 out 33.0 MB
  h16* bufS = (h16*)alloc(9000000);    // max: f3 squeeze 8.3 MB
  float* pooled = (float*)alloc(8 * 512 * sizeof(float));
  float* par1   = (float*)alloc(4 * 384 * sizeof(float));
  float* par2   = (float*)alloc(4 * 384 * sizeof(float));
  h16 *wsq[7], *we1[7], *we3[7];
  for (int i = 0; i < 7; ++i) {
    wsq[i] = (h16*)alloc((size_t)FC[i].sqp * FC[i].cin * sizeof(h16));
    we1[i] = (h16*)alloc((size_t)FC[i].ex * FC[i].sqp * sizeof(h16));
    we3[i] = (h16*)alloc((size_t)FC[i].ex * 9 * FC[i].sqp * sizeof(h16));
  }
  if (off > ws_size) return;

  // ---- weight repacks (fp32 -> padded f16) ----
  for (int i = 0; i < 7; ++i) {
    const int base = 3 + 6 * i;
    const float* sw  = (const float*)d_in[base + 0];
    const float* e1w = (const float*)d_in[base + 2];
    const float* e3w = (const float*)d_in[base + 4];
    long t0 = (long)FC[i].sqp * FC[i].cin;
    repack_w1x1<<<(int)((t0 + 255) / 256), 256, 0, stream>>>(sw, wsq[i], FC[i].sq, FC[i].cin, FC[i].sqp, FC[i].cin);
    long t1 = (long)FC[i].ex * FC[i].sqp;
    repack_w1x1<<<(int)((t1 + 255) / 256), 256, 0, stream>>>(e1w, we1[i], FC[i].ex, FC[i].sq, FC[i].ex, FC[i].sqp);
    long t2 = (long)FC[i].ex * 9 * FC[i].sqp;
    repack_w3x3<<<(int)((t2 + 255) / 256), 256, 0, stream>>>(e3w, we3[i], FC[i].ex, FC[i].sq, FC[i].ex, FC[i].sqp);
  }

  // ---- stem ----
  {
    long tot = 8L * 255 * 255;
    conv0_stem<<<(int)((tot + 255) / 256), 256, 0, stream>>>(x, c0w, c0b, bufA);
  }

  auto pool = [&](const h16* in, h16* out, int H, int W, int C) {
    const int Ho = (H - 3) / 2 + 1, Wo = (W - 3) / 2 + 1;
    long tot = 8L * Ho * Wo * C;
    maxpool3x3s2<<<(int)((tot + 255) / 256), 256, 0, stream>>>(in, out, 8, H, W, C, Ho, Wo);
  };

  auto fire = [&](int i, const h16* in, h16* outb, int H, int W) {
    const FireCfg& f = FC[i];
    const int base = 3 + 6 * i;
    const float* sb  = (const float*)d_in[base + 1];
    const float* e1b = (const float*)d_in[base + 3];
    const float* e3b = (const float*)d_in[base + 5];
    const int P = 8 * H * W;
    GemmArgs gs = { in,   wsq[i], sb,  bufS, P, f.sq, f.sqp, 0 };
    launch_gemm(f.cin, f.sqp, gs, stream);
    GemmArgs ge = { bufS, we1[i], e1b, outb, P, f.ex, f.ex, 0 };
    launch_gemm(f.sqp, 2 * f.ex, ge, stream);
    ConvArgs ce = { bufS, we3[i], e3b, outb, H, W, f.ex, f.ex, f.ex };
    launch_conv(f.sqp, 2 * f.ex, ce, stream);
  };

  // ---- backbone pipeline (ping-pong A/B, squeeze in S) ----
  pool(bufA, bufB, 255, 255, 64);     // -> [8][127][127][64]
  fire(0, bufB, bufA, 127, 127);      // f3  -> A (128ch)
  fire(1, bufA, bufB, 127, 127);      // f4  -> B
  pool(bufB, bufA, 127, 127, 128);    // -> [8][63][63][128]
  fire(2, bufA, bufB, 63, 63);        // f6  -> B (256ch)
  fire(3, bufB, bufA, 63, 63);        // f7  -> A
  pool(bufA, bufB, 63, 63, 256);      // -> [8][31][31][256]
  fire(4, bufB, bufA, 31, 31);        // f9  -> A (384ch)
  fire(5, bufA, bufB, 31, 31);        // f10 -> B
  fire(6, bufB, bufA, 31, 31);        // f11 -> A (512ch)

  // ---- head ----
  avgpool_global<<<8, 512, 0, stream>>>(bufA, pooled, 31 * 31, 512);
  fc_pair<<<(4 * 384 + 255) / 256, 256, 0, stream>>>(
      pooled, (const float*)d_in[45], (const float*)d_in[46],
      (const float*)d_in[49], (const float*)d_in[50], par1);
  fc_pair<<<(4 * 384 + 255) / 256, 256, 0, stream>>>(
      pooled, (const float*)d_in[47], (const float*)d_in[48],
      (const float*)d_in[51], (const float*)d_in[52], par2);

  // ---- curve filters + blend ----
  {
    dim3 grid(128, 3, 4);
    curve_apply<<<grid, 256, 0, stream>>>(
        x, par1, par2, (const float*)d_in[53], (const float*)d_in[54], (float*)d_out);
  }
}
